// GRUStage2_51436528337767
// MI455X (gfx1250) — matrix operand order repaired and checked by reading, not compiled, not run on hardware
//
#include <hip/hip_runtime.h>
#include <hip/hip_bf16.h>

// ---------------------------------------------------------------------------
// Problem constants (match reference)
// ---------------------------------------------------------------------------
#define TT   4096
#define INF_ 768
#define HH   512
#define H2   1024
#define H3   1536
#define VV   14
#define NLL  5
#define LL   1024

typedef __attribute__((ext_vector_type(16))) __bf16 v16bf;
typedef __attribute__((ext_vector_type(8)))  float  v8f;

union BFU { unsigned short u; __bf16 h; };

__device__ __forceinline__ float sigmoidf_(float x) { return 1.0f / (1.0f + __expf(-x)); }

__device__ __forceinline__ float block_reduce_sum(float v, float* red, int tid, int n) {
    red[tid] = v; __syncthreads();
    for (int off = n >> 1; off > 0; off >>= 1) {
        if (tid < off) red[tid] += red[tid + off];
        __syncthreads();
    }
    float r = red[0]; __syncthreads();
    return r;
}

__device__ __forceinline__ float block_reduce_max(float v, float* red, int tid, int n) {
    red[tid] = v; __syncthreads();
    for (int off = n >> 1; off > 0; off >>= 1) {
        if (tid < off) red[tid] = fmaxf(red[tid], red[tid + off]);
        __syncthreads();
    }
    float r = red[0]; __syncthreads();
    return r;
}

// ---------------------------------------------------------------------------
// fp32 -> bf16 (RNE) conversion
// ---------------------------------------------------------------------------
__global__ void __launch_bounds__(256) f32_to_bf16_kernel(const float* __restrict__ in,
                                                          unsigned short* __restrict__ out, int n) {
    int i = blockIdx.x * 256 + threadIdx.x;
    if (i < n) {
        unsigned int u = __float_as_uint(in[i]);
        out[i] = (unsigned short)((u + 0x7fffu + ((u >> 16) & 1u)) >> 16);
    }
}

// ---------------------------------------------------------------------------
// WMMA fragment loaders (contiguous k order, k = 16*(lane/16) + s, for both A and B).
// A 16x32 bf16: lanes 0-15 hold K+0..15 (V0-7) of row M=lane;
//               lanes 16-31 hold K+16..31.  -> two b128 loads.
// B 32x16 bf16: lanes 0-15 hold K+0..15 of col N=lane; lanes 16-31 K+16..31.
// ---------------------------------------------------------------------------
__device__ __forceinline__ v16bf load_fragA(const unsigned short* __restrict__ Ar,
                                            int kk, int ahalf) {
    v16bf f;
#pragma unroll
    for (int i = 0; i < 8; i++) {
        const int ka = kk + ahalf + (i << 1);
        unsigned int av = *(const unsigned int*)(Ar + ka);
        BFU a0, a1; a0.u = (unsigned short)(av & 0xffffu); a1.u = (unsigned short)(av >> 16);
        f[2 * i] = a0.h; f[2 * i + 1] = a1.h;
    }
    return f;
}

__device__ __forceinline__ v16bf load_fragB(const unsigned short* __restrict__ Wr,
                                            int kk, int bhalf) {
    v16bf f;
#pragma unroll
    for (int i = 0; i < 8; i++) {
        const int kb = kk + bhalf + (i << 1);
        unsigned int bv = *(const unsigned int*)(Wr + kb);
        BFU b0, b1; b0.u = (unsigned short)(bv & 0xffffu); b1.u = (unsigned short)(bv >> 16);
        f[2 * i] = b0.h; f[2 * i + 1] = b1.h;
    }
    return f;
}

// ---------------------------------------------------------------------------
// WMMA GEMM:  C[M,N] = A[M,K] * W[N,K]^T + bias[N]
// A, W bf16 row-major (K contiguous), C fp32.
// 2x2 register blocking: each wave owns a 32x32 C macro-tile
// (2 A frags x 2 B frags -> 4 v_wmma per K-step; 2x the FLOP/byte of 1x1).
// Requires M%32==0, N%32==0, K%32==0.
// ---------------------------------------------------------------------------
__global__ void __launch_bounds__(256) wmma_gemm_bias(const unsigned short* __restrict__ A,
                                                      const unsigned short* __restrict__ W,
                                                      const float* __restrict__ bias,
                                                      float* __restrict__ C,
                                                      int M, int N, int K) {
    const int lane    = threadIdx.x & 31;
    const int wave    = threadIdx.x >> 5;
    const int tilesN2 = N >> 5;
    const int tile    = blockIdx.x * 8 + wave;
    if (tile >= (M >> 5) * tilesN2) return;         // wave-uniform exit (EXEC stays full)

    const int tm = tile / tilesN2;
    const int tn = tile % tilesN2;
    const int r0 = tm * 32 + (lane & 15);
    const int c0 = tn * 32 + (lane & 15);
    const int ahalf = (lane >> 4) * 16;
    const int bhalf = (lane >> 4) * 16;

    const unsigned short* A0 = A + (size_t)r0 * K;
    const unsigned short* A1 = A0 + (size_t)16 * K;
    const unsigned short* W0 = W + (size_t)c0 * K;
    const unsigned short* W1 = W0 + (size_t)16 * K;

    v8f acc00 = {}, acc01 = {}, acc10 = {}, acc11 = {};
    for (int kk = 0; kk < K; kk += 32) {
        if (kk + 32 < K) {                          // global_prefetch next K block
            __builtin_prefetch(A0 + kk + 32, 0, 1);
            __builtin_prefetch(A1 + kk + 32, 0, 1);
            __builtin_prefetch(W0 + kk + 32, 0, 1);
            __builtin_prefetch(W1 + kk + 32, 0, 1);
        }
        v16bf fa0 = load_fragA(A0, kk, ahalf);
        v16bf fa1 = load_fragA(A1, kk, ahalf);
        v16bf fb0 = load_fragB(W0, kk, bhalf);
        v16bf fb1 = load_fragB(W1, kk, bhalf);
        acc00 = __builtin_amdgcn_wmma_f32_16x16x32_bf16(false, fa0, false, fb0, (short)0, acc00, false, false);
        acc01 = __builtin_amdgcn_wmma_f32_16x16x32_bf16(false, fa0, false, fb1, (short)0, acc01, false, false);
        acc10 = __builtin_amdgcn_wmma_f32_16x16x32_bf16(false, fa1, false, fb0, (short)0, acc10, false, false);
        acc11 = __builtin_amdgcn_wmma_f32_16x16x32_bf16(false, fa1, false, fb1, (short)0, acc11, false, false);
    }

    // D layout: VGPR r holds M = r (+8 for lane>=16); lane%16 gives N.
    const int n0  = tn * 32 + (lane & 15);
    const int n1  = n0 + 16;
    const int mb0 = tm * 32 + ((lane >> 4) << 3);
    const int mb1 = mb0 + 16;
    const float bv0 = bias ? bias[n0] : 0.0f;
    const float bv1 = bias ? bias[n1] : 0.0f;
#pragma unroll
    for (int r = 0; r < 8; r++) {
        C[(size_t)(mb0 + r) * N + n0] = acc00[r] + bv0;
        C[(size_t)(mb0 + r) * N + n1] = acc01[r] + bv1;
        C[(size_t)(mb1 + r) * N + n0] = acc10[r] + bv0;
        C[(size_t)(mb1 + r) * N + n1] = acc11[r] + bv1;
    }
}

// ---------------------------------------------------------------------------
// Persistent GRU scan: one block per direction (blockIdx.x: 0=fwd, 1=bwd).
// Gi (precomputed Wih@x+bih) in global, h in LDS, Whh streamed from L2.
// ---------------------------------------------------------------------------
__global__ void __launch_bounds__(512) gru_scan_kernel(const float* __restrict__ GiF,
                                                       const float* __restrict__ GiB,
                                                       const float* __restrict__ WhhF,
                                                       const float* __restrict__ WhhB,
                                                       const float* __restrict__ bhhF,
                                                       const float* __restrict__ bhhB,
                                                       float* __restrict__ Y,
                                                       float* __restrict__ hTf,
                                                       float* __restrict__ hTb) {
    __shared__ float hsh[HH];
    const int tid = threadIdx.x;
    const int dir = blockIdx.x;
    const float* Gi = dir ? GiB : GiF;
    const float* Wh = dir ? WhhB : WhhF;
    const float* bh = dir ? bhhB : bhhF;
    float* hT       = dir ? hTb  : hTf;
    const int colOff = dir ? HH : 0;

    hsh[tid] = 0.0f;
    __syncthreads();

    const float bR = bh[tid], bZ = bh[tid + HH], bN = bh[tid + 2 * HH];
    const float4* wr = (const float4*)(Wh + (size_t)tid * HH);
    const float4* wz = (const float4*)(Wh + (size_t)(tid + HH) * HH);
    const float4* wn = (const float4*)(Wh + (size_t)(tid + 2 * HH) * HH);
    const float4* h4 = (const float4*)hsh;

    for (int s = 0; s < TT; s++) {
        const int t = dir ? (TT - 1 - s) : s;
        float aR = bR, aZ = bZ, aN = bN;
#pragma unroll 4
        for (int k = 0; k < HH / 4; k++) {
            float4 hv = h4[k];
            float4 a = wr[k], b = wz[k], c = wn[k];
            aR += a.x * hv.x + a.y * hv.y + a.z * hv.z + a.w * hv.w;
            aZ += b.x * hv.x + b.y * hv.y + b.z * hv.z + b.w * hv.w;
            aN += c.x * hv.x + c.y * hv.y + c.z * hv.z + c.w * hv.w;
        }
        const float* gi = Gi + (size_t)t * H3;
        const float r = sigmoidf_(gi[tid] + aR);
        const float z = sigmoidf_(gi[tid + HH] + aZ);
        const float n = tanhf(gi[tid + 2 * HH] + r * aN);
        const float hprev = hsh[tid];
        const float hnew  = (1.0f - z) * n + z * hprev;
        __syncthreads();
        hsh[tid] = hnew;
        __syncthreads();
        Y[(size_t)t * H2 + colOff + tid] = hnew;
    }
    hT[tid] = hsh[tid];
}

// ---------------------------------------------------------------------------
// Row layernorm over C columns: one block (256 thr) per row.
// ---------------------------------------------------------------------------
__global__ void __launch_bounds__(256) ln_rows_kernel(const float* __restrict__ X,
                                                      const float* __restrict__ w,
                                                      const float* __restrict__ b,
                                                      float* __restrict__ Y, int C) {
    __shared__ float red[256];
    const int row = blockIdx.x, tid = threadIdx.x;
    const float* xr = X + (size_t)row * C;
    float s = 0.0f;
    for (int k = tid; k < C; k += 256) s += xr[k];
    const float m = block_reduce_sum(s, red, tid, 256) / (float)C;
    float v = 0.0f;
    for (int k = tid; k < C; k += 256) { float d = xr[k] - m; v += d * d; }
    const float var = block_reduce_sum(v, red, tid, 256) / (float)C;
    const float rs = rsqrtf(var + 1e-5f);
    float* yr = Y + (size_t)row * C;
    for (int k = tid; k < C; k += 256) yr[k] = (xr[k] - m) * rs * w[k] + b[k];
}

// ---------------------------------------------------------------------------
// Encoder head: h0[l] = LN(concat(hf[l],hb[l])) @ fcW^T + fcb   (5 x 512)
// ---------------------------------------------------------------------------
__global__ void __launch_bounds__(512) enc_head_kernel(const float* __restrict__ hf,
                                                       const float* __restrict__ hb,
                                                       const float* __restrict__ lnw,
                                                       const float* __restrict__ lnb,
                                                       const float* __restrict__ fcW,
                                                       const float* __restrict__ fcb,
                                                       float* __restrict__ h0) {
    __shared__ float buf[H2];
    __shared__ float red[512];
    const int tid = threadIdx.x;
    for (int l = 0; l < NLL; l++) {
        buf[tid]      = hf[l * HH + tid];
        buf[HH + tid] = hb[l * HH + tid];
        __syncthreads();
        const float m = block_reduce_sum(buf[tid] + buf[HH + tid], red, tid, 512) / (float)H2;
        const float d0 = buf[tid] - m, d1 = buf[HH + tid] - m;
        const float var = block_reduce_sum(d0 * d0 + d1 * d1, red, tid, 512) / (float)H2;
        const float rs = rsqrtf(var + 1e-5f);
        buf[tid]      = d0 * rs * lnw[tid] + lnb[tid];
        buf[HH + tid] = d1 * rs * lnw[HH + tid] + lnb[HH + tid];
        __syncthreads();
        float a = fcb[tid];
        const float4* w4 = (const float4*)(fcW + (size_t)tid * H2);
        const float4* x4 = (const float4*)buf;
        for (int k = 0; k < H2 / 4; k++) {
            float4 w = w4[k], x = x4[k];
            a += w.x * x.x + w.y * x.y + w.z * x.z + w.w * x.w;
        }
        h0[l * HH + tid] = a;
        __syncthreads();
    }
}

// ---------------------------------------------------------------------------
// Persistent decoder: 1024 sequential steps in one 1024-thread block.
// ---------------------------------------------------------------------------
__global__ void __launch_bounds__(1024) decoder_kernel(
    const int* __restrict__ ytok, const float* __restrict__ embd,
    const float* __restrict__ Wq, const float* __restrict__ bq,
    const float* __restrict__ keyM, const float* __restrict__ valM,
    const float* __restrict__ h0,
    const float* __restrict__ Wih0, const float* __restrict__ Whh0,
    const float* __restrict__ Wih, const float* __restrict__ Whh,
    const float* __restrict__ bih, const float* __restrict__ bhh,
    const float* __restrict__ lnh_w, const float* __restrict__ lnh_b,
    const float* __restrict__ lno_w, const float* __restrict__ lno_b,
    const float* __restrict__ fcW, const float* __restrict__ fcb,
    float* __restrict__ out) {
    __shared__ float h[NLL][HH];        // LN'd hidden carried across steps
    __shared__ float bufA[HH], bufB[HH];
    __shared__ float gin[H2];
    __shared__ float q[HH];
    __shared__ float att[TT];
    __shared__ float lnbuf[H3];
    __shared__ float red[1024];

    const int tid = threadIdx.x;
    if (tid < HH)
        for (int l = 0; l < NLL; l++) h[l][tid] = h0[l * HH + tid];
    __syncthreads();

    for (int step = 0; step < LL; step++) {
        const int tok = (step == 0) ? 13 : ytok[step - 1];

        // q = Wq @ h[NL-1] + bq
        if (tid < HH) {
            float a = bq[tid];
            const float4* w4 = (const float4*)(Wq + (size_t)tid * HH);
            const float4* s4 = (const float4*)h[NLL - 1];
            for (int k = 0; k < HH / 4; k++) {
                float4 w = w4[k], s = s4[k];
                a += w.x * s.x + w.y * s.y + w.z * s.z + w.w * s.w;
            }
            q[tid] = a;
        }
        __syncthreads();

        // scores = key @ q, softmax over T=4096 (4 rows per thread)
        float sc[4]; float mx = -1e30f;
        for (int i = 0; i < 4; i++) {
            const int t = tid + i * 1024;
            float a = 0.0f;
            const float4* k4 = (const float4*)(keyM + (size_t)t * HH);
            const float4* q4 = (const float4*)q;
            for (int k = 0; k < HH / 4; k++) {
                float4 w = k4[k], s = q4[k];
                a += w.x * s.x + w.y * s.y + w.z * s.z + w.w * s.w;
            }
            sc[i] = a; mx = fmaxf(mx, a);
        }
        const float gmax = block_reduce_max(mx, red, tid, 1024);
        float lsum = 0.0f;
        for (int i = 0; i < 4; i++) {
            const float e = __expf(sc[i] - gmax);
            att[tid + i * 1024] = e;
            lsum += e;
        }
        __syncthreads();
        const float ssum = block_reduce_sum(lsum, red, tid, 1024);

        // context c = att @ val / ssum ; gin = [embd[tok], c]
        if (tid < HH) {
            float a = 0.0f;
            for (int t = 0; t < TT; t++) a += att[t] * valM[(size_t)t * HH + tid];
            gin[HH + tid] = a / ssum;
            gin[tid]      = embd[tok * HH + tid];
        }
        __syncthreads();

        // 5 GRU layers
        float* cur = gin; int Kl = H2;
        for (int l = 0; l < NLL; l++) {
            const float* Wi = (l == 0) ? Wih0 : (Wih + (size_t)(l - 1) * H3 * HH);
            const float* Wh = (l == 0) ? Whh0 : (Whh + (size_t)(l - 1) * H3 * HH);
            const float* bi = bih + l * H3;
            const float* bh = bhh + l * H3;
            float* outb = (l & 1) ? bufB : bufA;
            if (tid < HH) {
                float gr = bi[tid], gz = bi[tid + HH], gn = bi[tid + 2 * HH];
                {
                    const float4* wr = (const float4*)(Wi + (size_t)tid * Kl);
                    const float4* wz = (const float4*)(Wi + (size_t)(tid + HH) * Kl);
                    const float4* wn = (const float4*)(Wi + (size_t)(tid + 2 * HH) * Kl);
                    const float4* c4 = (const float4*)cur;
                    for (int k = 0; k < Kl / 4; k++) {
                        float4 cv = c4[k];
                        float4 a = wr[k], b = wz[k], c = wn[k];
                        gr += a.x * cv.x + a.y * cv.y + a.z * cv.z + a.w * cv.w;
                        gz += b.x * cv.x + b.y * cv.y + b.z * cv.z + b.w * cv.w;
                        gn += c.x * cv.x + c.y * cv.y + c.z * cv.z + c.w * cv.w;
                    }
                }
                float hr = bh[tid], hz = bh[tid + HH], hg = bh[tid + 2 * HH];
                {
                    const float4* wr = (const float4*)(Wh + (size_t)tid * HH);
                    const float4* wz = (const float4*)(Wh + (size_t)(tid + HH) * HH);
                    const float4* wn = (const float4*)(Wh + (size_t)(tid + 2 * HH) * HH);
                    const float4* h4 = (const float4*)h[l];
                    for (int k = 0; k < HH / 4; k++) {
                        float4 hv = h4[k];
                        float4 a = wr[k], b = wz[k], c = wn[k];
                        hr += a.x * hv.x + a.y * hv.y + a.z * hv.z + a.w * hv.w;
                        hz += b.x * hv.x + b.y * hv.y + b.z * hv.z + b.w * hv.w;
                        hg += c.x * hv.x + c.y * hv.y + c.z * hv.z + c.w * hv.w;
                    }
                }
                const float r = sigmoidf_(gr + hr);
                const float z = sigmoidf_(gz + hz);
                const float n = tanhf(gn + r * hg);
                outb[tid] = (1.0f - z) * n + z * h[l][tid];
            }
            __syncthreads();
            // h[l] <- LN(outb) for next step (per-layer LN over H)
            const float m = block_reduce_sum((tid < HH) ? outb[tid] : 0.0f, red, tid, 1024) / (float)HH;
            float dv = 0.0f;
            if (tid < HH) { const float d = outb[tid] - m; dv = d * d; }
            const float var = block_reduce_sum(dv, red, tid, 1024) / (float)HH;
            const float rs = rsqrtf(var + 1e-5f);
            if (tid < HH) h[l][tid] = (outb[tid] - m) * rs * lnh_w[tid] + lnh_b[tid];
            __syncthreads();
            cur = outb; Kl = HH;
        }

        // pre = [cur, gin] (3H); LN; logits
        float* fin = bufA;                      // l=4 wrote bufA
        lnbuf[tid] = (tid < HH) ? fin[tid] : gin[tid - HH];
        if (tid < HH) lnbuf[H2 + tid] = gin[HH + tid];
        __syncthreads();
        float lv = lnbuf[tid] + ((tid < HH) ? lnbuf[H2 + tid] : 0.0f);
        const float m = block_reduce_sum(lv, red, tid, 1024) / (float)H3;
        const float d0 = lnbuf[tid] - m;
        const float d1 = (tid < HH) ? (lnbuf[H2 + tid] - m) : 0.0f;
        const float var = block_reduce_sum(d0 * d0 + d1 * d1, red, tid, 1024) / (float)H3;
        const float rs = rsqrtf(var + 1e-5f);
        lnbuf[tid] = d0 * rs * lno_w[tid] + lno_b[tid];
        if (tid < HH) lnbuf[H2 + tid] = d1 * rs * lno_w[H2 + tid] + lno_b[H2 + tid];
        __syncthreads();
        if (tid < VV) {
            float a = fcb[tid];
            const float* w = fcW + (size_t)tid * H3;
            for (int k = 0; k < H3; k++) a += w[k] * lnbuf[k];
            out[(size_t)step * VV + tid] = a;
        }
        __syncthreads();
    }
}

// ---------------------------------------------------------------------------
// Host orchestration
// ---------------------------------------------------------------------------
extern "C" void kernel_launch(void* const* d_in, const int* in_sizes, int n_in,
                              void* d_out, int out_size, void* d_ws, size_t ws_size,
                              hipStream_t stream) {
    (void)in_sizes; (void)n_in; (void)out_size; (void)ws_size;

    const float* x        = (const float*)d_in[0];
    const int*   y        = (const int*)  d_in[1];
    const float* embd     = (const float*)d_in[2];
    const float* Wq       = (const float*)d_in[3];
    const float* bq       = (const float*)d_in[4];
    const float* Wk       = (const float*)d_in[5];
    const float* bk       = (const float*)d_in[6];
    const float* Wv       = (const float*)d_in[7];
    const float* bv       = (const float*)d_in[8];
    const float* enc_Wih0 = (const float*)d_in[9];
    const float* enc_Whh0 = (const float*)d_in[10];
    const float* enc_bih0 = (const float*)d_in[11];
    const float* enc_bhh0 = (const float*)d_in[12];
    const float* enc_Wih  = (const float*)d_in[13];
    const float* enc_Whh  = (const float*)d_in[14];
    const float* enc_bih  = (const float*)d_in[15];
    const float* enc_bhh  = (const float*)d_in[16];
    const float* enc_fc_W = (const float*)d_in[17];
    const float* enc_fc_b = (const float*)d_in[18];
    const float* enc_lnh_w= (const float*)d_in[19];
    const float* enc_lnh_b= (const float*)d_in[20];
    const float* enc_ln_w = (const float*)d_in[21];
    const float* enc_ln_b = (const float*)d_in[22];
    const float* dec_Wih0 = (const float*)d_in[23];
    const float* dec_Whh0 = (const float*)d_in[24];
    const float* dec_Wih  = (const float*)d_in[25];
    const float* dec_Whh  = (const float*)d_in[26];
    const float* dec_bih  = (const float*)d_in[27];
    const float* dec_bhh  = (const float*)d_in[28];
    const float* dec_lnh_w= (const float*)d_in[29];
    const float* dec_lnh_b= (const float*)d_in[30];
    const float* ln_out_w = (const float*)d_in[31];
    const float* ln_out_b = (const float*)d_in[32];
    const float* fc_out_W = (const float*)d_in[33];
    const float* fc_out_b = (const float*)d_in[34];
    float* out = (float*)d_out;

    // Workspace carve (aligned 256B)
    size_t off = 0;
    char* base = (char*)d_ws;
    auto carve = [&](size_t bytes) -> void* {
        void* r = base + off;
        off += (bytes + 255) & ~(size_t)255;
        return r;
    };
    unsigned short* Abf  = (unsigned short*)carve((size_t)TT * H2 * 2);        // activations bf16
    unsigned short* WbfF = (unsigned short*)carve((size_t)H3 * H2 * 2);        // Wih fwd bf16
    unsigned short* WbfB = (unsigned short*)carve((size_t)H3 * H2 * 2);        // Wih bwd bf16
    float* GiF  = (float*)carve((size_t)TT * H3 * 4);
    float* GiB  = (float*)carve((size_t)TT * H3 * 4);
    float* Ybuf = (float*)carve((size_t)TT * H2 * 4);
    float* hf   = (float*)carve((size_t)NLL * HH * 4);
    float* hb   = (float*)carve((size_t)NLL * HH * 4);
    float* h0   = (float*)carve((size_t)NLL * HH * 4);
    unsigned short* Wkbf = (unsigned short*)carve((size_t)HH * H2 * 2);
    unsigned short* Wvbf = (unsigned short*)carve((size_t)HH * H2 * 2);
    // Reuse large buffers after the scans finish:
    float* enc_out = GiF;                         // [T, 2H] fp32 (16.8MB <= 25.2MB)
    float* keyM    = GiB;                         // [T, H]
    float* valM    = GiB + (size_t)TT * HH;       // [T, H]

    auto conv = [&](const float* src, unsigned short* dst, int n) {
        f32_to_bf16_kernel<<<(n + 255) / 256, 256, 0, stream>>>(src, dst, n);
    };
    auto gemm = [&](const unsigned short* A, const unsigned short* W, const float* bias,
                    float* C, int M, int N, int K) {
        int tiles = (M / 32) * (N / 32);          // 32x32 macro-tile per wave
        wmma_gemm_bias<<<(tiles + 7) / 8, 256, 0, stream>>>(A, W, bias, C, M, N, K);
    };

    // ---- Encoder: 5 bidirectional GRU layers ----
    conv(x, Abf, TT * INF_);
    for (int l = 0; l < NLL; l++) {
        const int K = (l == 0) ? INF_ : H2;
        const float* WiF = (l == 0) ? enc_Wih0
                                    : enc_Wih + (size_t)((l - 1) * 2 + 0) * H3 * H2;
        const float* WiB = (l == 0) ? (enc_Wih0 + (size_t)H3 * INF_)
                                    : enc_Wih + (size_t)((l - 1) * 2 + 1) * H3 * H2;
        const float* WhF = (l == 0) ? enc_Whh0
                                    : enc_Whh + (size_t)((l - 1) * 2 + 0) * H3 * HH;
        const float* WhB = (l == 0) ? (enc_Whh0 + (size_t)H3 * HH)
                                    : enc_Whh + (size_t)((l - 1) * 2 + 1) * H3 * HH;
        const float* biF = (l == 0) ? enc_bih0 : enc_bih + (size_t)((l - 1) * 2 + 0) * H3;
        const float* biB = (l == 0) ? (enc_bih0 + H3) : enc_bih + (size_t)((l - 1) * 2 + 1) * H3;
        const float* bhF = (l == 0) ? enc_bhh0 : enc_bhh + (size_t)((l - 1) * 2 + 0) * H3;
        const float* bhB = (l == 0) ? (enc_bhh0 + H3) : enc_bhh + (size_t)((l - 1) * 2 + 1) * H3;

        conv(WiF, WbfF, H3 * K);
        conv(WiB, WbfB, H3 * K);
        gemm(Abf, WbfF, biF, GiF, TT, H3, K);          // Gi_fwd = x @ Wih_f^T + bih_f
        gemm(Abf, WbfB, biB, GiB, TT, H3, K);          // Gi_bwd
        gru_scan_kernel<<<2, 512, 0, stream>>>(GiF, GiB, WhF, WhB, bhF, bhB,
                                               Ybuf, hf + l * HH, hb + l * HH);
        conv(Ybuf, Abf, TT * H2);                      // next layer's A (bf16)
    }

    // ---- Encoder head + attention K/V (WMMA GEMMs) ----
    enc_head_kernel<<<1, 512, 0, stream>>>(hf, hb, enc_lnh_w, enc_lnh_b,
                                           enc_fc_W, enc_fc_b, h0);
    ln_rows_kernel<<<TT, 256, 0, stream>>>(Ybuf, enc_ln_w, enc_ln_b, enc_out, H2);
    conv(enc_out, Abf, TT * H2);
    conv(Wk, Wkbf, HH * H2);
    conv(Wv, Wvbf, HH * H2);
    gemm(Abf, Wkbf, bk, keyM, TT, HH, H2);             // key_mat = enc_out @ Wk^T + bk
    gemm(Abf, Wvbf, bv, valM, TT, HH, H2);             // val_mat

    // ---- Decoder: 1024 sequential attention+GRU steps ----
    decoder_kernel<<<1, 1024, 0, stream>>>(y, embd, Wq, bq, keyM, valM, h0,
                                           dec_Wih0, dec_Whh0, dec_Wih, dec_Whh,
                                           dec_bih, dec_bhh, dec_lnh_w, dec_lnh_b,
                                           ln_out_w, ln_out_b, fc_out_W, fc_out_b, out);
}
